// PDP_Net_Model_70557722739187
// MI455X (gfx1250) — matrix operand order repaired and checked by reading, not compiled, not run on hardware
//
#include <hip/hip_runtime.h>
#include <hip/hip_bf16.h>

// DGCNN (EdgeConv x4 + 512->1024 conv) for MI455X / gfx1250.
// GEMMs on v_wmma_f32_16x16x32_bf16; features staged as (B, N, Kpad) bf16 so
// every WMMA operand is fetched with contiguous global_load_b128s.
// Edge conv rewritten as (W1-W2)*x_ctr + W2*x_nbr (gather-add epilogue).

#define BATCH 8
#define NPTS  2048
#define KNN   20
#define NGROUP 8
#define EPSBN 1e-5f

typedef __attribute__((ext_vector_type(16))) __bf16 v16bf;
typedef __attribute__((ext_vector_type(8)))  __bf16 v8bf;
typedef __attribute__((ext_vector_type(8)))  float  v8f;

union BfTile { v16bf v; v8bf h[2]; };

static __device__ __forceinline__ float lrelu(float v) { return v > 0.f ? v : 0.2f * v; }

// ---------------- small utility kernels ----------------

__global__ void k_zero(float* __restrict__ p, int n) {
  int t = blockIdx.x * blockDim.x + threadIdx.x;
  if (t < n) p[t] = 0.f;
}

// squared norms per point: x (B,C,N) -> sq (B,N)
__global__ void k_sqnorm(const float* __restrict__ x, float* __restrict__ sq,
                         int C, int Nn, int total) {
  int t = blockIdx.x * blockDim.x + threadIdx.x;
  if (t >= total) return;
  int b = t / Nn, n = t % Nn;
  float s = 0.f;
  for (int c = 0; c < C; ++c) {
    float v = x[((long)b * C + c) * Nn + n];
    s += v * v;
  }
  sq[t] = s;
}

// features (B,C,N) f32 -> (B,N,Ktot) bf16 with channel offset coff;
// channels c in [C,Kspan) write zero padding.
__global__ void k_cvt(const float* __restrict__ x, __bf16* __restrict__ out,
                      int C, int Nn, int Kspan, int Ktot, int coff, int total) {
  int t = blockIdx.x * blockDim.x + threadIdx.x;
  if (t >= total) return;
  int c = t % Kspan;
  int n = (t / Kspan) % Nn;
  int b = t / (Kspan * Nn);
  float v = (c < C) ? x[((long)b * C + c) * Nn + n] : 0.f;
  out[((long)b * Nn + n) * Ktot + coff + c] = (__bf16)v;
}

// plain f32 -> bf16 (row-major matrix)
__global__ void k_cvt_mat(const float* __restrict__ w, __bf16* __restrict__ out, int total) {
  int t = blockIdx.x * blockDim.x + threadIdx.x;
  if (t < total) out[t] = (__bf16)w[t];
}

// weights (O,2C) -> A1=(W1-W2), A2=W2 as bf16 (O,Kpad), zero-padded in K
__global__ void k_prep_w(const float* __restrict__ w, __bf16* __restrict__ A1,
                         __bf16* __restrict__ A2, int O, int C, int Kpad, int total) {
  int t = blockIdx.x * blockDim.x + threadIdx.x;
  if (t >= total) return;
  int k = t % Kpad, o = t / Kpad;
  float w1 = (k < C) ? w[o * 2 * C + k] : 0.f;
  float w2 = (k < C) ? w[o * 2 * C + C + k] : 0.f;
  A1[(long)o * Kpad + k] = (__bf16)(w1 - w2);
  A2[(long)o * Kpad + k] = (__bf16)w2;
}

// ---------------- WMMA GEMM kernels ----------------
// bf16 operand layouts as loaded here (wave32), with K-contiguous storage:
//  A 16x32:  a[0..7]  = Arow[k0 + 16*half + 0..7]  (16B run)
//            a[8..15] = Arow[k0 + 16*half + 8..15]
//  B 32x16:  b[0..15] = Brow[k0 + 16*half + 0..15] (32B run, Brow = column n's K-run)
//  C/D 16x16 f32: vgpr r holds m = r + 8*half, lane%16 = n
// Each wave computes a 16x64 strip: one A tile shared by 4 B tiles.

// Gram: pd[n,m] = 2 * (x x^T)[n,m] - sq[n] - sq[m]; xT is (N, Kpad) bf16.
__global__ void k_gram_wmma(const __bf16* __restrict__ xT, const float* __restrict__ sq,
                            float* __restrict__ pd, int Nn, int Kpad) {
  const int wave = blockIdx.x * (blockDim.x >> 5) + (threadIdx.x >> 5);
  const int tilesM = Nn >> 4, tilesQ = Nn >> 6;
  const int tm = wave % tilesM, tq = wave / tilesM;
  if (tq >= tilesQ) return;
  const int lane = threadIdx.x & 31;
  const int half = lane >> 4, q = lane & 15;
  const int mrow = tm * 16 + q;
  const int ncol = tq * 64 + q;
  const __bf16* Arow = xT + (long)mrow * Kpad;
  const __bf16* Brow0 = xT + (long)(ncol +  0) * Kpad;
  const __bf16* Brow1 = xT + (long)(ncol + 16) * Kpad;
  const __bf16* Brow2 = xT + (long)(ncol + 32) * Kpad;
  const __bf16* Brow3 = xT + (long)(ncol + 48) * Kpad;
  v8f z = {};
  v8f acc0 = z, acc1 = z, acc2 = z, acc3 = z;
  for (int k0 = 0; k0 < Kpad; k0 += 32) {
    BfTile a, b0, b1, b2, b3;
    const v8bf* pa = (const v8bf*)(Arow + k0);
    a.h[0] = pa[2*half]; a.h[1] = pa[2*half + 1];
    const v8bf* pb0 = (const v8bf*)(Brow0 + k0);
    const v8bf* pb1 = (const v8bf*)(Brow1 + k0);
    const v8bf* pb2 = (const v8bf*)(Brow2 + k0);
    const v8bf* pb3 = (const v8bf*)(Brow3 + k0);
    b0.h[0] = pb0[2*half]; b0.h[1] = pb0[2*half + 1];
    b1.h[0] = pb1[2*half]; b1.h[1] = pb1[2*half + 1];
    b2.h[0] = pb2[2*half]; b2.h[1] = pb2[2*half + 1];
    b3.h[0] = pb3[2*half]; b3.h[1] = pb3[2*half + 1];
    acc0 = __builtin_amdgcn_wmma_f32_16x16x32_bf16(false, a.v, false, b0.v, (short)0, acc0, false, false);
    acc1 = __builtin_amdgcn_wmma_f32_16x16x32_bf16(false, a.v, false, b1.v, (short)0, acc1, false, false);
    acc2 = __builtin_amdgcn_wmma_f32_16x16x32_bf16(false, a.v, false, b2.v, (short)0, acc2, false, false);
    acc3 = __builtin_amdgcn_wmma_f32_16x16x32_bf16(false, a.v, false, b3.v, (short)0, acc3, false, false);
  }
  const v8f* accs[4] = {&acc0, &acc1, &acc2, &acc3};
#pragma unroll
  for (int j = 0; j < 4; ++j) {
    const int n = ncol + 16 * j;
    const float sqn = sq[n];
#pragma unroll
    for (int r = 0; r < 8; ++r) {
      const int m = tm * 16 + r + 8 * half;
      pd[(long)m * Nn + n] = 2.f * (*accs[j])[r] - sq[m] - sqn;
    }
  }
}

// Generic: out(m,n) = sum_k A[m,k] * BT[n,k]; batched over blockIdx.y.
__global__ void k_gemm_wmma(const __bf16* __restrict__ A, const __bf16* __restrict__ BT,
                            float* __restrict__ out, int M, int Nn, int Kpad,
                            long bStrideB, long bStrideO, long so_m, long so_n) {
  const int b = blockIdx.y;
  const __bf16* Bp = BT + (long)b * bStrideB;   // (Nn, Kpad) row-major
  float* Op = out + (long)b * bStrideO;
  const int wave = blockIdx.x * (blockDim.x >> 5) + (threadIdx.x >> 5);
  const int tilesM = M >> 4, tilesQ = Nn >> 6;
  const int tm = wave % tilesM, tq = wave / tilesM;
  if (tq >= tilesQ) return;
  const int lane = threadIdx.x & 31;
  const int half = lane >> 4, q = lane & 15;
  const int mrow = tm * 16 + q;
  const int ncol = tq * 64 + q;
  const __bf16* Arow = A + (long)mrow * Kpad;
  const __bf16* Brow0 = Bp + (long)(ncol +  0) * Kpad;
  const __bf16* Brow1 = Bp + (long)(ncol + 16) * Kpad;
  const __bf16* Brow2 = Bp + (long)(ncol + 32) * Kpad;
  const __bf16* Brow3 = Bp + (long)(ncol + 48) * Kpad;
  v8f z = {};
  v8f acc0 = z, acc1 = z, acc2 = z, acc3 = z;
  for (int k0 = 0; k0 < Kpad; k0 += 32) {
    __builtin_prefetch(Brow0 + k0 + 32, 0, 1);  // global_prefetch_b8
    BfTile a, b0, b1, b2, b3;
    const v8bf* pa = (const v8bf*)(Arow + k0);
    a.h[0] = pa[2*half]; a.h[1] = pa[2*half + 1];
    const v8bf* pb0 = (const v8bf*)(Brow0 + k0);
    const v8bf* pb1 = (const v8bf*)(Brow1 + k0);
    const v8bf* pb2 = (const v8bf*)(Brow2 + k0);
    const v8bf* pb3 = (const v8bf*)(Brow3 + k0);
    b0.h[0] = pb0[2*half]; b0.h[1] = pb0[2*half + 1];
    b1.h[0] = pb1[2*half]; b1.h[1] = pb1[2*half + 1];
    b2.h[0] = pb2[2*half]; b2.h[1] = pb2[2*half + 1];
    b3.h[0] = pb3[2*half]; b3.h[1] = pb3[2*half + 1];
    acc0 = __builtin_amdgcn_wmma_f32_16x16x32_bf16(false, a.v, false, b0.v, (short)0, acc0, false, false);
    acc1 = __builtin_amdgcn_wmma_f32_16x16x32_bf16(false, a.v, false, b1.v, (short)0, acc1, false, false);
    acc2 = __builtin_amdgcn_wmma_f32_16x16x32_bf16(false, a.v, false, b2.v, (short)0, acc2, false, false);
    acc3 = __builtin_amdgcn_wmma_f32_16x16x32_bf16(false, a.v, false, b3.v, (short)0, acc3, false, false);
  }
  const v8f* accs[4] = {&acc0, &acc1, &acc2, &acc3};
#pragma unroll
  for (int j = 0; j < 4; ++j) {
    const int n = ncol + 16 * j;
#pragma unroll
    for (int r = 0; r < 8; ++r) {
      const int m = tm * 16 + r + 8 * half;
      Op[(long)m * so_m + (long)n * so_n] = (*accs[j])[r];
    }
  }
}

// ---------------- top-k (one wave32 per row, register-cached) ----------------

__global__ void k_topk(const float* __restrict__ pd, int* __restrict__ idxo) {
  const int row = blockIdx.x * (blockDim.x >> 5) + (threadIdx.x >> 5);
  const int lane = threadIdx.x & 31;
  const float* r = pd + (long)row * NPTS;
  float vals[NPTS / 32];
#pragma unroll
  for (int i = 0; i < NPTS / 32; ++i) vals[i] = r[lane + i * 32];
  for (int t = 0; t < KNN; ++t) {
    float best = -3.4e38f; int bi = 0;
#pragma unroll
    for (int i = 0; i < NPTS / 32; ++i)
      if (vals[i] > best) { best = vals[i]; bi = i; }
    float bv = best;
    int bidx = lane + bi * 32;
    for (int off = 16; off > 0; off >>= 1) {
      float ov = __shfl_xor(bv, off, 32);
      int   oi = __shfl_xor(bidx, off, 32);
      if (ov > bv || (ov == bv && oi < bidx)) { bv = ov; bidx = oi; }
    }
    if (lane == 0) idxo[row * KNN + t] = bidx;
    if ((bidx & 31) == lane) vals[bidx >> 5] = -3.4e38f;  // knockout
  }
}

// ---------------- edge conv epilogue ----------------
// y[b,o,n,k] = u[b,n,o] + v[b,idx[b,n,k],o]; accumulate sum/sumsq per o and
// per-(b,n,o) max over k. u, v, Mx stored (B,N,O) so o is coalesced.
__global__ void k_edge_reduce(const float* __restrict__ u, const float* __restrict__ v,
                              const int* __restrict__ nbr, float* __restrict__ Mx,
                              float* __restrict__ s1g, float* __restrict__ s2g,
                              int O, int Nn) {
  const int b = blockIdx.y;
  const int n0 = blockIdx.x * NGROUP;
  const int o = threadIdx.x;
  if (o >= O) return;
  const float* ub = u + (long)b * Nn * O;
  const float* vb = v + (long)b * Nn * O;
  float s1 = 0.f, s2 = 0.f;
  for (int dn = 0; dn < NGROUP; ++dn) {
    const int n = n0 + dn;
    const float yc = ub[(long)n * O + o];
    const int* ix = nbr + ((long)b * Nn + n) * KNN;
    float mx = -3.4e38f;
    for (int k = 0; k < KNN; ++k) {
      const int j = ix[k];
      const float y = yc + vb[(long)j * O + o];
      s1 += y; s2 += y * y;
      mx = fmaxf(mx, y);
    }
    Mx[((long)b * Nn + n) * O + o] = mx;
  }
  atomicAdd(&s1g[o], s1);   // global_atomic_add_f32
  atomicAdd(&s2g[o], s2);
}

// BN + LeakyReLU on the per-(b,n,o) max (commutes: gamma>0, monotone),
// writing features as (B,O,N) for the next layer.
__global__ void k_bn_max_fin(const float* __restrict__ Mx, const float* __restrict__ s1,
                             const float* __restrict__ s2, const float* __restrict__ g,
                             const float* __restrict__ be, float* __restrict__ xout,
                             int O, int Nn, float invcnt, int total) {
  int t = blockIdx.x * blockDim.x + threadIdx.x;
  if (t >= total) return;
  const int n = t % Nn;
  const int o = (t / Nn) % O;
  const int b = t / (Nn * O);
  const float mean = s1[o] * invcnt;
  const float var  = s2[o] * invcnt - mean * mean;
  const float val = (Mx[((long)b * Nn + n) * O + o] - mean) * rsqrtf(var + EPSBN) * g[o] + be[o];
  xout[((long)b * O + o) * Nn + n] = lrelu(val);
}

// ---------------- layer 5 stats + finalize ----------------

__global__ void k_stats5(const float* __restrict__ y, float* __restrict__ s1,
                         float* __restrict__ s2, int O, int Nn) {
  const int o = blockIdx.x;
  float a = 0.f, qq = 0.f;
  const int tot = BATCH * Nn;
  for (int i = threadIdx.x; i < tot; i += blockDim.x) {
    const int bb = i / Nn, n = i % Nn;
    const float v = y[((long)bb * O + o) * Nn + n];
    a += v; qq += v * v;
  }
  __shared__ float sa[256], sb[256];
  sa[threadIdx.x] = a; sb[threadIdx.x] = qq;
  __syncthreads();
  for (int s = 128; s > 0; s >>= 1) {
    if (threadIdx.x < s) { sa[threadIdx.x] += sa[threadIdx.x + s]; sb[threadIdx.x] += sb[threadIdx.x + s]; }
    __syncthreads();
  }
  if (threadIdx.x == 0) { s1[o] = sa[0]; s2[o] = sb[0]; }
}

__global__ void k_final5(const float* __restrict__ y, const float* __restrict__ s1,
                         const float* __restrict__ s2, const float* __restrict__ g,
                         const float* __restrict__ be, float* __restrict__ out,
                         int O, int Nn) {
  const int o = blockIdx.x % O, b = blockIdx.x / O;
  float mx = -3.4e38f;
  for (int n = threadIdx.x; n < Nn; n += blockDim.x)
    mx = fmaxf(mx, y[((long)b * O + o) * Nn + n]);
  __shared__ float sm[256];
  sm[threadIdx.x] = mx; __syncthreads();
  for (int s = 128; s > 0; s >>= 1) {
    if (threadIdx.x < s) sm[threadIdx.x] = fmaxf(sm[threadIdx.x], sm[threadIdx.x + s]);
    __syncthreads();
  }
  if (threadIdx.x == 0) {
    const float invc = 1.f / (float)(BATCH * Nn);
    const float mean = s1[o] * invc;
    const float var  = s2[o] * invc - mean * mean;
    out[b * O + o] = lrelu((sm[0] - mean) * rsqrtf(var + EPSBN) * g[o] + be[o]);
  }
}

// ---------------- host driver ----------------

extern "C" void kernel_launch(void* const* d_in, const int* in_sizes, int n_in,
                              void* d_out, int out_size, void* d_ws, size_t ws_size,
                              hipStream_t stream) {
  (void)in_sizes; (void)n_in; (void)out_size; (void)ws_size;
  const float* x  = (const float*)d_in[0];
  const float* W[5]  = {(const float*)d_in[1], (const float*)d_in[4], (const float*)d_in[7],
                        (const float*)d_in[10], (const float*)d_in[13]};
  const float* G[5]  = {(const float*)d_in[2], (const float*)d_in[5], (const float*)d_in[8],
                        (const float*)d_in[11], (const float*)d_in[14]};
  const float* Be[5] = {(const float*)d_in[3], (const float*)d_in[6], (const float*)d_in[9],
                        (const float*)d_in[12], (const float*)d_in[15]};

  // workspace carve-up (256B aligned)
  char* base = (char*)d_ws;
  size_t off = 0;
  auto take = [&](size_t bytes) -> void* {
    void* p = base + off; off += (bytes + 255) & ~(size_t)255; return p;
  };
  float*  pd   = (float*)take((size_t)NPTS * NPTS * 4);          // one batch at a time
  int*    idxb = (int*)take((size_t)BATCH * NPTS * KNN * 4);
  float*  sq   = (float*)take((size_t)BATCH * NPTS * 4);
  __bf16* xb   = (__bf16*)take((size_t)BATCH * NPTS * 128 * 2);  // (B,N,Kpad), Kpad<=128
  __bf16* A1   = (__bf16*)take((size_t)256 * 128 * 2);
  __bf16* A2   = (__bf16*)take((size_t)256 * 128 * 2);
  float*  u    = (float*)take((size_t)BATCH * NPTS * 256 * 4);
  float*  v    = (float*)take((size_t)BATCH * NPTS * 256 * 4);
  float*  Mx   = (float*)take((size_t)BATCH * NPTS * 256 * 4);
  float*  s1   = (float*)take((size_t)2 * 1024 * 4);
  float*  s2   = s1 + 1024;
  float*  x1b  = (float*)take((size_t)BATCH * 64  * NPTS * 4);
  float*  x2b  = (float*)take((size_t)BATCH * 64  * NPTS * 4);
  float*  x3b  = (float*)take((size_t)BATCH * 128 * NPTS * 4);
  float*  x4b  = (float*)take((size_t)BATCH * 256 * NPTS * 4);
  __bf16* xcb  = (__bf16*)take((size_t)BATCH * NPTS * 512 * 2); // (B,N,512)
  __bf16* w5b  = (__bf16*)take((size_t)1024 * 512 * 2);
  float*  y5   = (float*)take((size_t)BATCH * 1024 * NPTS * 4);

  auto cdiv = [](long a, long b) { return (int)((a + b - 1) / b); };
  auto round32 = [](int c) { return (c + 31) & ~31; };

  auto gemm = [&](const __bf16* Am, const __bf16* Bm, float* out, int M, int Nn, int Kpad,
                  long bsB, long bsO, long so_m, long so_n) {
    const int tiles = (M / 16) * (Nn / 64);
    k_gemm_wmma<<<dim3(cdiv(tiles, 8), BATCH), 256, 0, stream>>>(
        Am, Bm, out, M, Nn, Kpad, bsB, bsO, so_m, so_n);
  };

  auto run_layer = [&](const float* xin, int C, int O, const float* w, const float* g,
                       const float* be, float* xout) {
    const int Kpad = round32(C);
    int total = BATCH * NPTS;
    k_sqnorm<<<cdiv(total, 256), 256, 0, stream>>>(xin, sq, C, NPTS, total);
    total = BATCH * Kpad * NPTS;
    k_cvt<<<cdiv(total, 256), 256, 0, stream>>>(xin, xb, C, NPTS, Kpad, Kpad, 0, total);
    total = O * Kpad;
    k_prep_w<<<cdiv(total, 256), 256, 0, stream>>>(w, A1, A2, O, C, Kpad, total);
    // knn: gram + topk, batch at a time (pd reused; stream order serializes)
    const int gtiles = (NPTS / 16) * (NPTS / 64);
    for (int b = 0; b < BATCH; ++b) {
      k_gram_wmma<<<cdiv(gtiles, 8), 256, 0, stream>>>(
          xb + (long)b * NPTS * Kpad, sq + b * NPTS, pd, NPTS, Kpad);
      k_topk<<<NPTS / 8, 256, 0, stream>>>(pd, idxb + (long)b * NPTS * KNN);
    }
    // u = (W1-W2)*x, v = W2*x, stored (B,N,O)
    gemm(A1, xb, u, O, NPTS, Kpad, (long)NPTS * Kpad, (long)NPTS * O, 1, O);
    gemm(A2, xb, v, O, NPTS, Kpad, (long)NPTS * Kpad, (long)NPTS * O, 1, O);
    k_zero<<<cdiv(2 * 1024, 256), 256, 0, stream>>>(s1, 2 * 1024);
    k_edge_reduce<<<dim3(NPTS / NGROUP, BATCH), O, 0, stream>>>(u, v, idxb, Mx, s1, s2, O, NPTS);
    const long tot = (long)BATCH * O * NPTS;
    k_bn_max_fin<<<cdiv(tot, 256), 256, 0, stream>>>(
        Mx, s1, s2, g, be, xout, O, NPTS, 1.f / ((float)BATCH * NPTS * KNN), (int)tot);
  };

  run_layer(x,   3,   64,  W[0], G[0], Be[0], x1b);
  run_layer(x1b, 64,  64,  W[1], G[1], Be[1], x2b);
  run_layer(x2b, 64,  128, W[2], G[2], Be[2], x3b);
  run_layer(x3b, 128, 256, W[3], G[3], Be[3], x4b);

  // concat [x1,x2,x3,x4] -> xcb (B,N,512) bf16
  int total = BATCH * 64 * NPTS;
  k_cvt<<<cdiv(total, 256), 256, 0, stream>>>(x1b, xcb, 64, NPTS, 64, 512, 0, total);
  k_cvt<<<cdiv(total, 256), 256, 0, stream>>>(x2b, xcb, 64, NPTS, 64, 512, 64, total);
  total = BATCH * 128 * NPTS;
  k_cvt<<<cdiv(total, 256), 256, 0, stream>>>(x3b, xcb, 128, NPTS, 128, 512, 128, total);
  total = BATCH * 256 * NPTS;
  k_cvt<<<cdiv(total, 256), 256, 0, stream>>>(x4b, xcb, 256, NPTS, 256, 512, 256, total);
  k_cvt_mat<<<cdiv(1024 * 512, 256), 256, 0, stream>>>(W[4], w5b, 1024 * 512);

  // y5 = w5 * xc, stored (B,1024,N)
  gemm(w5b, xcb, y5, 1024, NPTS, 512, (long)NPTS * 512, (long)1024 * NPTS, NPTS, 1);
  k_stats5<<<1024, 256, 0, stream>>>(y5, s1, s2, 1024, NPTS);
  k_final5<<<BATCH * 1024, 256, 0, stream>>>(y5, s1, s2, G[4], Be[4], (float*)d_out, 1024, NPTS);
}
